// RotaryAttention_16381005267440
// MI455X (gfx1250) — matrix operand order repaired and checked by reading, not compiled, not run on hardware
//
#include <hip/hip_runtime.h>
#include <hip/hip_bf16.h>

// ---------------------------------------------------------------------------
// RotaryAttention (GQA + RoPE) for MI455X / gfx1250, wave32, WMMA bf16.
// B=4, S=512, D=4096, H=32, G=8, HD=128, REP=4.
// All large GEMMs use v_wmma_f32_16x16x32_bf16 (fp32 accumulate).
// ---------------------------------------------------------------------------

#define BB 4
#define SS 512
#define DD 4096
#define HH 32
#define GG 8
#define HDD 128
#define REPP 4

typedef __attribute__((ext_vector_type(16))) __bf16 bf16x16;
typedef __attribute__((ext_vector_type(8)))  float  floatx8;

union BF16Frag {
    bf16x16 v;
    uint4   q[2];
};

__device__ __forceinline__ __bf16 f2bf(float f) {
    // round-to-nearest-even fp32 -> bf16
    unsigned u = __builtin_bit_cast(unsigned, f);
    unsigned r = (u + 0x7FFFu + ((u >> 16) & 1u)) >> 16;
    unsigned short h = (unsigned short)r;
    return __builtin_bit_cast(__bf16, h);
}

// ---------------------------------------------------------------------------
// Elementwise fp32 -> bf16 convert
// ---------------------------------------------------------------------------
__global__ void cvt_f32_bf16(const float* __restrict__ in, __bf16* __restrict__ out, long n) {
    long i = (long)blockIdx.x * blockDim.x + threadIdx.x;
    if (i < n) out[i] = f2bf(in[i]);
}

// ---------------------------------------------------------------------------
// Tiled transpose + convert: wt[c][r] = bf16(w[r][c]).  rows, cols % 32 == 0.
// block (32,8), 32x32 tile via LDS.
// ---------------------------------------------------------------------------
__global__ void transpose_cvt(const float* __restrict__ w, __bf16* __restrict__ wt,
                              int rows, int cols) {
    __shared__ float tile[32][33];
    int c0 = blockIdx.x * 32, r0 = blockIdx.y * 32;
    int tx = threadIdx.x, ty = threadIdx.y;
#pragma unroll
    for (int i = 0; i < 4; i++) {
        int r = r0 + ty + i * 8;
        tile[ty + i * 8][tx] = w[(long)r * cols + c0 + tx];
    }
    __syncthreads();
#pragma unroll
    for (int i = 0; i < 4; i++) {
        int c = c0 + ty + i * 8;
        wt[(long)c * rows + r0 + tx] = f2bf(tile[tx][ty + i * 8]);
    }
}

// ---------------------------------------------------------------------------
// Batched WMMA GEMM:   C[m][n] = sum_k A[m][k] * Bt[n][k]  (+ bias[n])
//   A  : bf16, row-major M x K      (per-batch stride a_bstride elements)
//   Bt : bf16, row-major N x K      (batch index = batch / bt_div, GQA share)
//   C  : fp32, row stride c_rstride, batch offset =
//            (batch / c_split) * c_ostride + (batch % c_split) * c_istride
// Block: 256 threads = 8 waves. Wave tile 16(M) x 64(N); block tile 64 x 128.
// A/B fragment loads are contiguous 16B loads; A and B share one contiguous k order:
//   A: lane holds row (lane&15), elems at k+half*16 .. +15
//   B: lane holds col (lane&15), elems at k+half*16 .. +15
// ---------------------------------------------------------------------------
__global__ void gemm_bf16_wmma(const __bf16* __restrict__ A,
                               const __bf16* __restrict__ Bt,
                               float* __restrict__ C,
                               const float* __restrict__ bias,
                               int M, int N, int K,
                               long a_bstride, long bt_bstride, int bt_div,
                               int c_split, long c_ostride, long c_istride,
                               int c_rstride) {
    int batch = blockIdx.z;
    const __bf16* Ab = A  + (long)batch * a_bstride;
    const __bf16* Bb = Bt + (long)(batch / bt_div) * bt_bstride;
    float* Cb = C + (long)(batch / c_split) * c_ostride
                  + (long)(batch % c_split) * c_istride;

    int wave = threadIdx.x >> 5;
    int lane = threadIdx.x & 31;
    int wm = wave & 3, wn = wave >> 2;
    int m0 = blockIdx.x * 64  + wm * 16;
    int n0 = blockIdx.y * 128 + wn * 64;
    if (m0 >= M || n0 >= N) return;

    int row  = lane & 15;
    int half = lane >> 4;

    const __bf16* arow = Ab + (long)(m0 + row) * K + half * 16;
    const __bf16* brow0 = Bb + (long)(n0 +  0 + row) * K + half * 16;
    const __bf16* brow1 = Bb + (long)(n0 + 16 + row) * K + half * 16;
    const __bf16* brow2 = Bb + (long)(n0 + 32 + row) * K + half * 16;
    const __bf16* brow3 = Bb + (long)(n0 + 48 + row) * K + half * 16;

    floatx8 acc0, acc1, acc2, acc3;
#pragma unroll
    for (int i = 0; i < 8; i++) { acc0[i] = 0.f; acc1[i] = 0.f; acc2[i] = 0.f; acc3[i] = 0.f; }

    for (int k0 = 0; k0 < K; k0 += 32) {
        // prefetch ~8 iterations ahead (speculative; OOB silently dropped)
        __builtin_prefetch(arow  + k0 + 256, 0, 0);
        __builtin_prefetch(brow0 + k0 + 256, 0, 0);
        __builtin_prefetch(brow2 + k0 + 256, 0, 0);

        BF16Frag a;
        a.q[0] = *reinterpret_cast<const uint4*>(arow + k0);
        a.q[1] = *reinterpret_cast<const uint4*>(arow + k0 + 8);

        BF16Frag b0, b1, b2, b3;
        b0.q[0] = *reinterpret_cast<const uint4*>(brow0 + k0);
        b0.q[1] = *reinterpret_cast<const uint4*>(brow0 + k0 + 8);
        b1.q[0] = *reinterpret_cast<const uint4*>(brow1 + k0);
        b1.q[1] = *reinterpret_cast<const uint4*>(brow1 + k0 + 8);
        b2.q[0] = *reinterpret_cast<const uint4*>(brow2 + k0);
        b2.q[1] = *reinterpret_cast<const uint4*>(brow2 + k0 + 8);
        b3.q[0] = *reinterpret_cast<const uint4*>(brow3 + k0);
        b3.q[1] = *reinterpret_cast<const uint4*>(brow3 + k0 + 8);

        acc0 = __builtin_amdgcn_wmma_f32_16x16x32_bf16(false, a.v, false, b0.v,
                                                       (short)0, acc0, false, false);
        acc1 = __builtin_amdgcn_wmma_f32_16x16x32_bf16(false, a.v, false, b1.v,
                                                       (short)0, acc1, false, false);
        acc2 = __builtin_amdgcn_wmma_f32_16x16x32_bf16(false, a.v, false, b2.v,
                                                       (short)0, acc2, false, false);
        acc3 = __builtin_amdgcn_wmma_f32_16x16x32_bf16(false, a.v, false, b3.v,
                                                       (short)0, acc3, false, false);
    }

    // C layout (16x16 f32): VGPR r, lanes 0-15 -> M = r, lanes 16-31 -> M = r+8
    floatx8 accs[4] = {acc0, acc1, acc2, acc3};
#pragma unroll
    for (int t = 0; t < 4; t++) {
        int n = n0 + t * 16 + row;
        float bv = bias ? bias[n] : 0.0f;
#pragma unroll
        for (int r = 0; r < 8; r++) {
            int m = m0 + r + half * 8;
            Cb[(long)m * c_rstride + n] = accs[t][r] + bv;
        }
    }
}

// ---------------------------------------------------------------------------
// RoPE for Q: reads qraw fp32 (b,s,D), writes qbf (b,h,s,hd) with 1/HD folded
// (reference applies HD^-0.5 to q and again to logits => q * 1/HD here).
// One thread per (b,s,h,j), j in [0,64): handles the (j, j+64) rotation pair.
// ---------------------------------------------------------------------------
__global__ void rope_q(const float* __restrict__ qraw, __bf16* __restrict__ qbf) {
    long i = (long)blockIdx.x * blockDim.x + threadIdx.x;
    int j = i & 63; long t = i >> 6;
    int h = (int)(t % HH); t /= HH;
    int s = (int)(t % SS); int b = (int)(t / SS);

    long src = ((long)(b * SS + s)) * DD + h * HDD + j;
    float x1 = qraw[src];
    float x2 = qraw[src + 64];

    float inv = __powf(10000.0f, -(float)j * (1.0f / 64.0f));
    float ang = (float)s * inv;
    float cs = __cosf(ang), sn = __sinf(ang);
    const float sc = 1.0f / (float)HDD;

    float o1 = (x1 * cs - x2 * sn) * sc;
    float o2 = (x2 * cs + x1 * sn) * sc;

    long dst = ((long)(b * HH + h) * SS + s) * HDD + j;
    qbf[dst]      = f2bf(o1);
    qbf[dst + 64] = f2bf(o2);
}

// RoPE for K: kraw fp32 (b,s,G*HD) -> kbf (b,g,s,hd), no extra scaling.
__global__ void rope_k(const float* __restrict__ kraw, __bf16* __restrict__ kbf) {
    long i = (long)blockIdx.x * blockDim.x + threadIdx.x;
    int j = i & 63; long t = i >> 6;
    int g = (int)(t % GG); t /= GG;
    int s = (int)(t % SS); int b = (int)(t / SS);

    long src = ((long)(b * SS + s)) * (GG * HDD) + g * HDD + j;
    float x1 = kraw[src];
    float x2 = kraw[src + 64];

    float inv = __powf(10000.0f, -(float)j * (1.0f / 64.0f));
    float ang = (float)s * inv;
    float cs = __cosf(ang), sn = __sinf(ang);

    long dst = ((long)(b * GG + g) * SS + s) * HDD + j;
    kbf[dst]      = f2bf(x1 * cs - x2 * sn);
    kbf[dst + 64] = f2bf(x2 * cs + x1 * sn);
}

// V: transpose+convert vraw (b,s,G*HD) -> vT (b,g,hd,s)
__global__ void pack_v(const float* __restrict__ vraw, __bf16* __restrict__ vT) {
    long i = (long)blockIdx.x * blockDim.x + threadIdx.x;
    int hd = i & 127; long t = i >> 7;
    int g = (int)(t % GG); t /= GG;
    int s = (int)(t % SS); int b = (int)(t / SS);

    float v = vraw[((long)(b * SS + s)) * (GG * HDD) + g * HDD + hd];
    vT[((long)(b * GG + g) * HDD + hd) * SS + s] = f2bf(v);
}

// ---------------------------------------------------------------------------
// Softmax: one wave (32 lanes) per 512-wide row, fp32 in -> bf16 out.
// 8 rows per 256-thread block.
// ---------------------------------------------------------------------------
__global__ void softmax_rows(const float* __restrict__ logits, __bf16* __restrict__ attn) {
    int wave = threadIdx.x >> 5, lane = threadIdx.x & 31;
    long row = (long)blockIdx.x * 8 + wave;
    const float* src = logits + row * SS;
    __bf16* dst = attn + row * SS;

    float vals[16];
    float m = -3.4e38f;
#pragma unroll
    for (int t = 0; t < 16; t++) {
        vals[t] = src[lane + t * 32];
        m = fmaxf(m, vals[t]);
    }
#pragma unroll
    for (int off = 16; off > 0; off >>= 1) m = fmaxf(m, __shfl_xor(m, off, 32));

    float sum = 0.f;
#pragma unroll
    for (int t = 0; t < 16; t++) { vals[t] = __expf(vals[t] - m); sum += vals[t]; }
#pragma unroll
    for (int off = 16; off > 0; off >>= 1) sum += __shfl_xor(sum, off, 32);

    float invs = 1.0f / sum;
#pragma unroll
    for (int t = 0; t < 16; t++) dst[lane + t * 32] = f2bf(vals[t] * invs);
}

// ---------------------------------------------------------------------------
// Host-side orchestration
// ---------------------------------------------------------------------------
extern "C" void kernel_launch(void* const* d_in, const int* in_sizes, int n_in,
                              void* d_out, int out_size, void* d_ws, size_t ws_size,
                              hipStream_t stream) {
    const float* x  = (const float*)d_in[0];
    const float* wq = (const float*)d_in[1];
    const float* bq = (const float*)d_in[2];
    const float* wk = (const float*)d_in[3];
    const float* bk = (const float*)d_in[4];
    const float* wv = (const float*)d_in[5];
    const float* bv = (const float*)d_in[6];
    const float* wo = (const float*)d_in[7];
    const float* bo = (const float*)d_in[8];
    float* out = (float*)d_out;

    const long MROWS = (long)BB * SS;          // 2048
    const long KV_N  = (long)GG * HDD;         // 1024

    // workspace carve-up (all sizes multiples of 256B)
    char* ws = (char*)d_ws;
    size_t off = 0;
    auto carve = [&](size_t bytes) -> void* {
        void* p = ws + off;
        off += (bytes + 255) & ~(size_t)255;
        return p;
    };
    __bf16* xbf   = (__bf16*)carve(MROWS * DD * 2);            // 16 MB
    __bf16* wqT   = (__bf16*)carve((size_t)DD * DD * 2);       // 32 MB
    __bf16* wkT   = (__bf16*)carve((size_t)KV_N * DD * 2);     //  8 MB
    __bf16* wvT   = (__bf16*)carve((size_t)KV_N * DD * 2);     //  8 MB
    __bf16* woT   = (__bf16*)carve((size_t)DD * DD * 2);       // 32 MB
    float*  qraw  = (float*)carve(MROWS * DD * 4);             // 32 MB
    float*  kraw  = (float*)carve(MROWS * KV_N * 4);           //  8 MB
    float*  vraw  = (float*)carve(MROWS * KV_N * 4);           //  8 MB
    __bf16* qbf   = (__bf16*)carve((size_t)BB * HH * SS * HDD * 2); // 16 MB
    __bf16* kbf   = (__bf16*)carve((size_t)BB * GG * SS * HDD * 2); //  4 MB
    __bf16* vT    = (__bf16*)carve((size_t)BB * GG * HDD * SS * 2); //  4 MB
    float*  logit = (float*)carve((size_t)BB * HH * SS * SS * 4);   // 128 MB
    __bf16* attn  = (__bf16*)carve((size_t)BB * HH * SS * SS * 2);  // 64 MB
    float*  ctxf  = (float*)carve(MROWS * DD * 4);             // 32 MB
    __bf16* ctxbf = (__bf16*)carve(MROWS * DD * 2);            // 16 MB
    (void)ws_size;

    // 1) convert x, transpose+convert weights to bf16 N x K
    {
        long n = MROWS * DD;
        cvt_f32_bf16<<<(n + 255) / 256, 256, 0, stream>>>(x, xbf, n);
        dim3 tb(32, 8);
        transpose_cvt<<<dim3(DD / 32, DD / 32), tb, 0, stream>>>(wq, wqT, DD, DD);
        transpose_cvt<<<dim3(KV_N / 32, DD / 32), tb, 0, stream>>>(wk, wkT, DD, (int)KV_N);
        transpose_cvt<<<dim3(KV_N / 32, DD / 32), tb, 0, stream>>>(wv, wvT, DD, (int)KV_N);
        transpose_cvt<<<dim3(DD / 32, DD / 32), tb, 0, stream>>>(wo, woT, DD, DD);
    }

    // 2) Q/K/V projections (WMMA bf16, fp32 out)
    gemm_bf16_wmma<<<dim3(MROWS / 64, DD / 128, 1), 256, 0, stream>>>(
        xbf, wqT, qraw, bq, (int)MROWS, DD, DD, 0, 0, 1, 1, 0, 0, DD);
    gemm_bf16_wmma<<<dim3(MROWS / 64, KV_N / 128, 1), 256, 0, stream>>>(
        xbf, wkT, kraw, bk, (int)MROWS, (int)KV_N, DD, 0, 0, 1, 1, 0, 0, (int)KV_N);
    gemm_bf16_wmma<<<dim3(MROWS / 64, KV_N / 128, 1), 256, 0, stream>>>(
        xbf, wvT, vraw, bv, (int)MROWS, (int)KV_N, DD, 0, 0, 1, 1, 0, 0, (int)KV_N);

    // 3) RoPE (+scale folding) and V transpose
    {
        long nq = (long)BB * SS * HH * 64;
        rope_q<<<(nq + 255) / 256, 256, 0, stream>>>(qraw, qbf);
        long nk = (long)BB * SS * GG * 64;
        rope_k<<<(nk + 255) / 256, 256, 0, stream>>>(kraw, kbf);
        long nv = (long)BB * SS * GG * HDD;
        pack_v<<<(nv + 255) / 256, 256, 0, stream>>>(vraw, vT);
    }

    // 4) logits[b,h] = q[b,h] @ k[b,h/REP]^T  (scale already folded into q)
    gemm_bf16_wmma<<<dim3(SS / 64, SS / 128, BB * HH), 256, 0, stream>>>(
        qbf, kbf, logit, nullptr, SS, SS, HDD,
        (long)SS * HDD, (long)SS * HDD, REPP,
        1, (long)SS * SS, 0, SS);

    // 5) softmax rows -> bf16 probabilities
    {
        long rows = (long)BB * HH * SS;
        softmax_rows<<<rows / 8, 256, 0, stream>>>(logit, attn);
    }

    // 6) ctx[b, s, h*HD + d] = attn[b,h] @ v[b, h/REP]   (C strided into (b,s,D))
    gemm_bf16_wmma<<<dim3(SS / 64, 1, BB * HH), 256, 0, stream>>>(
        attn, vT, ctxf, nullptr, SS, HDD, SS,
        (long)SS * SS, (long)HDD * SS, REPP,
        HH, (long)SS * DD, (long)HDD, DD);

    // 7) final projection: out = ctx @ wo + bo
    {
        long n = MROWS * DD;
        cvt_f32_bf16<<<(n + 255) / 256, 256, 0, stream>>>(ctxf, ctxbf, n);
    }
    gemm_bf16_wmma<<<dim3(MROWS / 64, DD / 128, 1), 256, 0, stream>>>(
        ctxbf, woT, out, bo, (int)MROWS, DD, DD, 0, 0, 1, 1, 0, 0, DD);

    (void)in_sizes; (void)n_in; (void)out_size;
}
